// DGCNN_13657996001491
// MI455X (gfx1250) — matrix operand order repaired and checked by reading, not compiled, not run on hardware
//
#include <hip/hip_runtime.h>
#include <hip/hip_bf16.h>

// ---------------------------------------------------------------------------
// CDNA5 (gfx1250) DGCNN forward.  wave32; WMMA bf16 16x16x32 for the two
// GEMM-shaped hot spots (KNN Gram matrices, Wf feature GEMM).
// ---------------------------------------------------------------------------

typedef __attribute__((ext_vector_type(16))) __bf16 v16bf;
typedef __attribute__((ext_vector_type(8)))  float  v8f;

#define LRELU_SLOPE 0.2f
#define BN_EPS 1e-5f
#define NPTS 4096
#define NB 8

__device__ __forceinline__ unsigned short f2bf(float f) {
  unsigned u = __float_as_uint(f);
  unsigned r = u + 0x7FFFu + ((u >> 16) & 1u);   // round to nearest even
  return (unsigned short)(r >> 16);
}

__device__ __forceinline__ float lrelu(float v) {
  return v > 0.f ? v : LRELU_SLOPE * v;
}

__device__ __forceinline__ void atomicMaxF(float* addr, float v) {
  unsigned* ua = (unsigned*)addr;
  unsigned old = *ua;
  while (__uint_as_float(old) < v) {
    unsigned assumed = old;
    old = atomicCAS(ua, assumed, __float_as_uint(v));
    if (old == assumed) break;
  }
}

// ---------------------------------------------------------------------------
// Generic fill
// ---------------------------------------------------------------------------
__global__ void fill_kernel(float* p, float v, int n) {
  int i = blockIdx.x * blockDim.x + threadIdx.x;
  if (i < n) p[i] = v;
}

// ---------------------------------------------------------------------------
// Pack fp32 (strided) matrix into CDNA5 WMMA bf16 fragment layout.
//   A fragment (16x32 bf16, ISA 7.12.2): lane L: m=L%16, half=L/16;
//     VGPR v (pairs): k = 2*(v&3) + (e&1) + 8*half + 16*(v>>2)
//   B fragment (32x16 bf16, same k order as A): lane L: n=L%16;
//     k = 2*(v&3) + (e&1) + 8*half + 16*(v>>2)
// One block = one (b, tile, kchunk); 32 lanes; each lane writes 32 bytes.
// ---------------------------------------------------------------------------
__global__ void pack_kernel(const float* __restrict__ src,
                            unsigned long long bS, unsigned long long cS,
                            unsigned long long nS, int C, int nTiles, int nkc,
                            int isB, uint4* __restrict__ dst) {
  int lane = threadIdx.x;
  int kc = blockIdx.x, t = blockIdx.y, b = blockIdx.z;
  int p = t * 16 + (lane & 15);
  int half = lane >> 4;
  unsigned u[8];
#pragma unroll
  for (int v = 0; v < 8; ++v) {
    unsigned short h01[2];
#pragma unroll
    for (int h = 0; h < 2; ++h) {
      int k;
      if (isB) k = ((v & 3) << 1) + h + (half << 3) + ((v >> 2) << 4);
      else     k = ((v & 3) << 1) + h + (half << 3) + ((v >> 2) << 4);
      int c = kc * 32 + k;
      float val = (c < C)
          ? src[(unsigned long long)b * bS + (unsigned long long)c * cS +
                (unsigned long long)p * nS]
          : 0.f;
      h01[h] = f2bf(val);
    }
    u[v] = (unsigned)h01[0] | ((unsigned)h01[1] << 16);
  }
  unsigned long long base =
      ((((unsigned long long)b * nTiles + t) * nkc + kc) * 32 + lane) * 2ull;
  dst[base]     = make_uint4(u[0], u[1], u[2], u[3]);
  dst[base + 1] = make_uint4(u[4], u[5], u[6], u[7]);
}

// ---------------------------------------------------------------------------
// Per-point squared norms
// ---------------------------------------------------------------------------
__global__ void x2_kernel(const float* __restrict__ src, unsigned long long bS,
                          int C, float* __restrict__ x2, int N) {
  int i = blockIdx.x * blockDim.x + threadIdx.x;
  if (i >= NB * N) return;
  int b = i / N, n = i - b * N;
  float s = 0.f;
  for (int c = 0; c < C; ++c) {
    float v = src[(unsigned long long)b * bS + (unsigned long long)c * N + n];
    s += v * v;
  }
  x2[i] = s;
}

// ---------------------------------------------------------------------------
// Fused KNN: Gram matrix via v_wmma_f32_16x16x32_bf16, distance, top-9.
// Block = 1 wave, handles 16 rows x all 4096 cols. Lanes 0..15 own one row
// each for the top-9 maintenance (register select-chains, no scratch).
// ---------------------------------------------------------------------------
template <int NKC>
__global__ void knn_kernel(const v16bf* __restrict__ pa,
                           const v16bf* __restrict__ pb,
                           const float* __restrict__ x2,
                           int* __restrict__ knn, int nTiles, int N) {
  __shared__ float lds[16][16];
  int lane = threadIdx.x;
  int i = blockIdx.x, b = blockIdx.y;
  const float* x2b = x2 + b * N;

  v16bf a[NKC];
#pragma unroll
  for (int kc = 0; kc < NKC; ++kc)
    a[kc] = pa[((unsigned long long)(b * nTiles + i) * NKC + kc) * 32 + lane];

  float bd[9];
  int bi[9];
#pragma unroll
  for (int t = 0; t < 9; ++t) { bd[t] = 3.0e38f; bi[t] = 0; }
  float worst = 3.0e38f;
  int wslot = 0;

  int nc = lane & 15, half = lane >> 4;

  for (int j = 0; j < nTiles; ++j) {
    v8f c = {};
#pragma unroll
    for (int kc = 0; kc < NKC; ++kc) {
      v16bf bb =
          pb[((unsigned long long)(b * nTiles + j) * NKC + kc) * 32 + lane];
      c = __builtin_amdgcn_wmma_f32_16x16x32_bf16(false, a[kc], false, bb,
                                                  (short)0, c, false, false);
    }
    if (j + 1 < nTiles)
      __builtin_prefetch(
          &pb[((unsigned long long)(b * nTiles + j + 1) * NKC) * 32 + lane], 0,
          1);
#pragma unroll
    for (int r = 0; r < 8; ++r) {
      int M = r + 8 * half;
      int grow = i * 16 + M, gcol = j * 16 + nc;
      float d = x2b[grow] + x2b[gcol] - 2.0f * c[r];
      if (grow == gcol) d += 1.0e10f;
      lds[M][nc] = d;
    }
    __syncthreads();
    if (lane < 16) {
#pragma unroll 4
      for (int col = 0; col < 16; ++col) {
        float d = lds[lane][col];
        int gcol = j * 16 + col;
        if (d < worst) {
#pragma unroll
          for (int t = 0; t < 9; ++t) {
            bool sel = (t == wslot);
            bd[t] = sel ? d : bd[t];
            bi[t] = sel ? gcol : bi[t];
          }
          worst = bd[0]; wslot = 0;
#pragma unroll
          for (int t = 1; t < 9; ++t)
            if (bd[t] > worst) { worst = bd[t]; wslot = t; }
        }
      }
    }
    __syncthreads();
  }
  if (lane < 16) {
    int g = i * 16 + lane;
#pragma unroll
    for (int t = 0; t < 9; ++t) knn[((unsigned long long)b * N + g) * 9 + t] = bi[t];
  }
}

// ---------------------------------------------------------------------------
// Edge conv: y[o,n,k] = Wa.xi + Wb.(xj-xi).  BN(axes 0,2,3) stats gathered in
// the same pass; since bn+lrelu is monotonic increasing, max over k commutes,
// so we store max_k(y) and normalize later.
// ---------------------------------------------------------------------------
#define ECH 8
__global__ void edge_kernel(const float* __restrict__ src,
                            unsigned long long bS, int C,
                            const float* __restrict__ W, int O,
                            const int* __restrict__ knn,
                            float* __restrict__ e, float* __restrict__ stats,
                            int N) {
  __shared__ float sxi[128];
  __shared__ float sxj[9][128];
  __shared__ int sidx[9];
  int tid = threadIdx.x, OT = blockDim.x;
  int b = blockIdx.y;
  int n0 = blockIdx.x * ECH;
  float lsum = 0.f, lsq = 0.f;

  for (int nn = 0; nn < ECH; ++nn) {
    int n = n0 + nn;
    if (tid < 9) sidx[tid] = knn[((unsigned long long)b * N + n) * 9 + tid];
    __syncthreads();
    for (int t = tid; t < 10 * C; t += OT) {
      int c = t % C, slot = t / C;
      if (slot == 0)
        sxi[c] = src[(unsigned long long)b * bS + (unsigned long long)c * N + n];
      else
        sxj[slot - 1][c] = src[(unsigned long long)b * bS +
                               (unsigned long long)c * N + sidx[slot - 1]];
    }
    __syncthreads();
    if (tid < O) {
      const float* Wa = W + (unsigned long long)tid * 2 * C;
      const float* Wb = Wa + C;
      float ai = 0.f;
      for (int c = 0; c < C; ++c) ai += (Wa[c] - Wb[c]) * sxi[c];
      float mx = -3.0e38f;
      for (int j = 0; j < 9; ++j) {
        float y = ai;
        for (int c = 0; c < C; ++c) y += Wb[c] * sxj[j][c];
        lsum += y; lsq += y * y;
        mx = fmaxf(mx, y);
      }
      e[((unsigned long long)b * O + tid) * N + n] = mx;
    }
    __syncthreads();
  }
  if (tid < O) {
    atomicAdd(&stats[tid], lsum);
    atomicAdd(&stats[O + tid], lsq);
  }
}

__global__ void bnfin_kernel(const float* __restrict__ stats,
                             float* __restrict__ ms, int O, float invCnt) {
  int o = blockIdx.x * blockDim.x + threadIdx.x;
  if (o >= O) return;
  float m = stats[o] * invCnt;
  float v = stats[O + o] * invCnt - m * m;
  ms[o] = m;
  ms[O + o] = rsqrtf(v + BN_EPS);
}

__global__ void applybn_kernel(float* __restrict__ e,
                               const float* __restrict__ ms, int O, int N,
                               int total) {
  int i = blockIdx.x * blockDim.x + threadIdx.x;
  if (i >= total) return;
  int o = (i / N) % O;
  e[i] = lrelu((e[i] - ms[o]) * ms[O + o]);
}

// ---------------------------------------------------------------------------
// Weighted (dilated, Gaussian-position-weighted) 1-D conv.
// ---------------------------------------------------------------------------
#define WCH 4
__global__ void wconv_kernel(const float* __restrict__ in, int C,
                             const float* __restrict__ coords,
                             const float* __restrict__ W,
                             const float* __restrict__ bias, int O, int K,
                             int dil, float inv2s2, float* __restrict__ feat,
                             int choff, int N) {
  __shared__ float sg[9];
  __shared__ int ss[9];
  int tid = threadIdx.x;
  int b = blockIdx.y;
  int n0 = blockIdx.x * WCH;
  int P = dil * (K / 2);
  for (int nn = 0; nn < WCH; ++nn) {
    int n = n0 + nn;
    if (tid < K) {
      int s = n + tid * dil - P;
      if (s >= 0 && s < N) {
        const float* cb = coords + (unsigned long long)b * 3 * N;
        float dx = cb[s] - cb[n];
        float dy = cb[N + s] - cb[N + n];
        float dz = cb[2 * N + s] - cb[2 * N + n];
        sg[tid] = expf(-(dx * dx + dy * dy + dz * dz) * inv2s2);
        ss[tid] = s;
      } else {
        sg[tid] = 0.f;
        ss[tid] = -1;
      }
    }
    __syncthreads();
    if (tid < O) {
      float acc = 0.f;
      for (int j = 0; j < K; ++j) {
        int s = ss[j];
        if (s < 0) continue;
        float dot = 0.f;
        for (int c = 0; c < C; ++c)
          dot += W[((unsigned long long)tid * C + c) * K + j] *
                 in[((unsigned long long)b * C + c) * N + s];
        acc += sg[j] * dot;
      }
      feat[((unsigned long long)b * 512 + choff + tid) * N + n] =
          acc + bias[tid];
    }
    __syncthreads();
  }
}

// ---------------------------------------------------------------------------
// Wf GEMM (1024x512 @ 512x4096 per batch) via WMMA bf16.
// mode 0: accumulate per-channel sum / sumsq (BN stats over axes 0,2).
// mode 1: normalize + lrelu, reduce max & sum over columns into pools.
// ---------------------------------------------------------------------------
__global__ void wf_gemm_kernel(const v16bf* __restrict__ wa,
                               const v16bf* __restrict__ fb, int mode,
                               const float* __restrict__ ms,
                               float* __restrict__ stats,
                               float* __restrict__ fmax,
                               float* __restrict__ fsum, int N) {
  const int NT = 256, NKC = 16;
  int lane = threadIdx.x;
  int j = blockIdx.x, i = blockIdx.y, b = blockIdx.z;
  v8f c = {};
#pragma unroll
  for (int kc = 0; kc < NKC; ++kc) {
    v16bf a = wa[((unsigned long long)i * NKC + kc) * 32 + lane];
    v16bf bb =
        fb[(((unsigned long long)b * NT + j) * NKC + kc) * 32 + lane];
    c = __builtin_amdgcn_wmma_f32_16x16x32_bf16(false, a, false, bb, (short)0,
                                                c, false, false);
  }
  int half = lane >> 4;
#pragma unroll
  for (int r = 0; r < 8; ++r) {
    int o = i * 16 + r + 8 * half;
    float v = c[r];
    if (mode == 0) {
      float s = v, sq = v * v;
      for (int m = 1; m < 16; m <<= 1) {
        s += __shfl_xor(s, m, 16);
        sq += __shfl_xor(sq, m, 16);
      }
      if ((lane & 15) == 0) {
        atomicAdd(&stats[o], s);
        atomicAdd(&stats[1024 + o], sq);
      }
    } else {
      float vv = lrelu((v - ms[o]) * ms[1024 + o]);
      float mx = vv, s = vv;
      for (int m = 1; m < 16; m <<= 1) {
        mx = fmaxf(mx, __shfl_xor(mx, m, 16));
        s += __shfl_xor(s, m, 16);
      }
      if ((lane & 15) == 0) {
        atomicMaxF(&fmax[b * 1024 + o], mx);
        atomicAdd(&fsum[b * 1024 + o], s);
      }
    }
  }
}

__global__ void pool_kernel(const float* __restrict__ fmax,
                            const float* __restrict__ fsum,
                            float* __restrict__ h, float invN) {
  int i = blockIdx.x * blockDim.x + threadIdx.x;
  if (i >= NB * 1024) return;
  int b = i >> 10, o = i & 1023;
  h[b * 2048 + o] = fmax[i];
  h[b * 2048 + 1024 + o] = fsum[i] * invN;
}

// ---------------------------------------------------------------------------
// Classifier MLP (tiny; VALU)
// ---------------------------------------------------------------------------
__global__ void dense_kernel(const float* __restrict__ in,
                             const float* __restrict__ W,
                             float* __restrict__ out, int CI, int O) {
  int i = blockIdx.x * blockDim.x + threadIdx.x;
  if (i >= NB * O) return;
  int b = i / O, o = i - b * O;
  float acc = 0.f;
  for (int c = 0; c < CI; ++c)
    acc += in[(unsigned long long)b * CI + c] * W[(unsigned long long)o * CI + c];
  out[b * O + o] = acc;
}

__global__ void cbn_kernel(float* __restrict__ t, int O) {
  int o = blockIdx.x * blockDim.x + threadIdx.x;
  if (o >= O) return;
  float s = 0.f, sq = 0.f;
  for (int b = 0; b < NB; ++b) {
    float v = t[b * O + o];
    s += v; sq += v * v;
  }
  float m = s * (1.0f / NB);
  float var = sq * (1.0f / NB) - m * m;
  float sc = rsqrtf(var + BN_EPS);
  for (int b = 0; b < NB; ++b)
    t[b * O + o] = lrelu((t[b * O + o] - m) * sc);
}

__global__ void densebias_kernel(const float* __restrict__ in,
                                 const float* __restrict__ W,
                                 const float* __restrict__ bias,
                                 float* __restrict__ out, int CI, int O) {
  int i = blockIdx.x * blockDim.x + threadIdx.x;
  if (i >= NB * O) return;
  int b = i / O, o = i - b * O;
  float acc = 0.f;
  for (int c = 0; c < CI; ++c)
    acc += in[(unsigned long long)b * CI + c] * W[(unsigned long long)o * CI + c];
  out[b * O + o] = acc + bias[o];
}

// ---------------------------------------------------------------------------
// Host orchestration
// ---------------------------------------------------------------------------
extern "C" void kernel_launch(void* const* d_in, const int* in_sizes, int n_in,
                              void* d_out, int out_size, void* d_ws,
                              size_t ws_size, hipStream_t stream) {
  (void)in_sizes; (void)n_in; (void)out_size; (void)ws_size;
  const int N = NPTS, B = NB;
  const int nTiles = N / 16;  // 256

  const float* x      = (const float*)d_in[0];
  const float* coords = (const float*)d_in[1];
  const float* We1 = (const float*)d_in[2];
  const float* We2 = (const float*)d_in[3];
  const float* We3 = (const float*)d_in[4];
  const float* We4 = (const float*)d_in[5];
  const float* W1 = (const float*)d_in[6];
  const float* b1 = (const float*)d_in[7];
  const float* W2 = (const float*)d_in[8];
  const float* b2 = (const float*)d_in[9];
  const float* W3 = (const float*)d_in[10];
  const float* b3 = (const float*)d_in[11];
  const float* W4 = (const float*)d_in[12];
  const float* b4 = (const float*)d_in[13];
  const float* Wf = (const float*)d_in[14];
  const float* Wc1 = (const float*)d_in[15];
  const float* Wc2 = (const float*)d_in[16];
  const float* Wc3 = (const float*)d_in[17];
  const float* bc3 = (const float*)d_in[18];

  size_t off = 0;
  auto alloc = [&](size_t bytes) -> void* {
    void* p = (char*)d_ws + off;
    off += (bytes + 255) & ~(size_t)255;
    return p;
  };
  float* feat  = (float*)alloc((size_t)B * 512 * N * 4);      // out1..out4
  float* ebuf  = (float*)alloc((size_t)B * 128 * N * 4);      // edge output
  uint4* packA = (uint4*)alloc((size_t)B * nTiles * 4 * 1024);
  uint4* packB = (uint4*)alloc((size_t)B * nTiles * 4 * 1024);
  float* x2b   = (float*)alloc((size_t)B * N * 4);
  int*   idxb  = (int*)alloc((size_t)B * N * 9 * 4);
  float* estats = (float*)alloc(2 * 128 * 4);
  float* ems    = (float*)alloc(2 * 128 * 4);
  uint4* wfA    = (uint4*)alloc((size_t)64 * 16 * 1024);
  uint4* featB  = (uint4*)alloc((size_t)B * nTiles * 16 * 1024);
  float* fstats = (float*)alloc(2048 * 4);
  float* fms    = (float*)alloc(2048 * 4);
  float* fmax   = (float*)alloc((size_t)B * 1024 * 4);
  float* fsum   = (float*)alloc((size_t)B * 1024 * 4);
  float* hbuf   = (float*)alloc((size_t)B * 2048 * 4);
  float* t1     = (float*)alloc((size_t)B * 512 * 4);
  float* t2     = (float*)alloc((size_t)B * 256 * 4);

  auto launch_knn = [&](int nkc) {
    dim3 kg(nTiles, B);
    if (nkc == 1)
      knn_kernel<1><<<kg, 32, 0, stream>>>((const v16bf*)packA,
                                           (const v16bf*)packB, x2b, idxb,
                                           nTiles, N);
    else if (nkc == 2)
      knn_kernel<2><<<kg, 32, 0, stream>>>((const v16bf*)packA,
                                           (const v16bf*)packB, x2b, idxb,
                                           nTiles, N);
    else
      knn_kernel<4><<<kg, 32, 0, stream>>>((const v16bf*)packA,
                                           (const v16bf*)packB, x2b, idxb,
                                           nTiles, N);
  };

  auto stage = [&](const float* src, unsigned long long bS, int C, int nkc,
                   const float* We, int Oe, const float* Wc, const float* bc,
                   int Oc, int K, int dil, float sigma, int choff) {
    // --- KNN ---
    dim3 pg(nkc, nTiles, B);
    pack_kernel<<<pg, 32, 0, stream>>>(src, bS, (unsigned long long)N, 1ull, C,
                                       nTiles, nkc, 0, packA);
    pack_kernel<<<pg, 32, 0, stream>>>(src, bS, (unsigned long long)N, 1ull, C,
                                       nTiles, nkc, 1, packB);
    x2_kernel<<<(B * N) / 256, 256, 0, stream>>>(src, bS, C, x2b, N);
    launch_knn(nkc);
    // --- edge conv + BN stats ---
    fill_kernel<<<1, 256, 0, stream>>>(estats, 0.f, 2 * Oe);
    dim3 eg(N / ECH, B);
    int OT = Oe < 32 ? 32 : Oe;
    edge_kernel<<<eg, OT, 0, stream>>>(src, bS, C, We, Oe, idxb, ebuf, estats,
                                       N);
    bnfin_kernel<<<1, 256, 0, stream>>>(estats, ems, Oe,
                                        1.0f / ((float)B * N * 9));
    int tot = B * Oe * N;
    applybn_kernel<<<(tot + 255) / 256, 256, 0, stream>>>(ebuf, ems, Oe, N,
                                                          tot);
    // --- weighted conv -> feat[choff .. choff+Oc) ---
    dim3 wg(N / WCH, B);
    wconv_kernel<<<wg, Oc, 0, stream>>>(ebuf, Oe, coords, Wc, bc, Oc, K, dil,
                                        1.0f / (2.0f * sigma * sigma), feat,
                                        choff, N);
  };

  stage(x, 3ull * N, 3, 1, We1, 3, W1, b1, 64, 9, 1, 0.02f, 0);
  stage(feat, 512ull * N, 64, 2, We2, 64, W2, b2, 64, 5, 2, 0.02f, 64);
  stage(feat + 64 * N, 512ull * N, 64, 2, We3, 64, W3, b3, 128, 5, 4, 0.04f,
        128);
  stage(feat + 128 * N, 512ull * N, 128, 4, We4, 128, W4, b4, 256, 5, 8, 0.08f,
        256);

  // --- Wf GEMM: f = lrelu(bn(Wf @ feat)), then max/mean pool over n ---
  pack_kernel<<<dim3(16, 64, 1), 32, 0, stream>>>(Wf, 0ull, 1ull, 512ull, 512,
                                                  64, 16, 0, wfA);
  pack_kernel<<<dim3(16, nTiles, B), 32, 0, stream>>>(
      feat, 512ull * N, (unsigned long long)N, 1ull, 512, nTiles, 16, 1, featB);
  fill_kernel<<<8, 256, 0, stream>>>(fstats, 0.f, 2048);
  wf_gemm_kernel<<<dim3(nTiles, 64, B), 32, 0, stream>>>(
      (const v16bf*)wfA, (const v16bf*)featB, 0, fms, fstats, fmax, fsum, N);
  bnfin_kernel<<<4, 256, 0, stream>>>(fstats, fms, 1024,
                                      1.0f / ((float)B * N));
  fill_kernel<<<32, 256, 0, stream>>>(fmax, -3.0e38f, B * 1024);
  fill_kernel<<<32, 256, 0, stream>>>(fsum, 0.f, B * 1024);
  wf_gemm_kernel<<<dim3(nTiles, 64, B), 32, 0, stream>>>(
      (const v16bf*)wfA, (const v16bf*)featB, 1, fms, fstats, fmax, fsum, N);
  pool_kernel<<<(B * 1024) / 256, 256, 0, stream>>>(fmax, fsum, hbuf,
                                                    1.0f / (float)N);

  // --- classifier MLP ---
  dense_kernel<<<(B * 512 + 255) / 256, 256, 0, stream>>>(hbuf, Wc1, t1, 2048,
                                                          512);
  cbn_kernel<<<2, 256, 0, stream>>>(t1, 512);
  dense_kernel<<<(B * 256 + 255) / 256, 256, 0, stream>>>(t1, Wc2, t2, 512,
                                                          256);
  cbn_kernel<<<1, 256, 0, stream>>>(t2, 256);
  densebias_kernel<<<(B * 40 + 255) / 256, 256, 0, stream>>>(
      t2, Wc3, bc3, (float*)d_out, 256, 40);
}
